// CrossModalAttention2_48576080118569
// MI455X (gfx1250) — compile-verified, matrix operand order checked by reading, not run on hardware
//
#include <hip/hip_runtime.h>
#include <math.h>

// ---------------------------------------------------------------------------
// CDNA5 / gfx1250 cross-modal attention block.  All matmuls via
// v_wmma_f32_16x16x32_bf16 (f32 accumulate).  wave32 throughout.
// Attention K-tiles streamed with GLOBAL_LOAD_ASYNC_TO_LDS_B128 (ASYNCcnt).
// ---------------------------------------------------------------------------

typedef __attribute__((ext_vector_type(16))) __bf16 v16bf;
typedef __attribute__((ext_vector_type(8)))  __bf16 v8bf;
typedef __attribute__((ext_vector_type(4)))  __bf16 v4bf;
typedef __attribute__((ext_vector_type(8)))  float  v8f;
typedef __attribute__((ext_vector_type(4)))  float  v4f;

#define E_DIM   512
#define H_HEADS 8
#define DH      64
#define S_LEN   1024
#define B_BATCH 8
#define M_TOT   (B_BATCH * S_LEN)   // 8192

// ---------------------------------------------------------------------------
// WMMA helpers
// ---------------------------------------------------------------------------
__device__ __forceinline__ v8f wmma_bf16(v16bf a, v16bf b, v8f c) {
  // 8 args: (neg_a, A, neg_b, B, c_mod, C, reuse_a, reuse_b)
  return __builtin_amdgcn_wmma_f32_16x16x32_bf16(false, a, false, b, (short)0, c,
                                                 false, false);
}

// Load one 16x32 bf16 operand tile in the ISA-documented A/B VGPR layout.
// Per-lane element i maps to K = i + (i>=8?8:0) + (hi?8:0), i.e. two
// contiguous 8-element (16 B) chunks at K-offsets {hi*8} and {16+hi*8}.
// Serves A tiles ([M][K] row-major, row = lane&15) and B tiles from a
// [N][K] row-major matrix (col = lane&15).  Works for global or LDS ptrs.
__device__ __forceinline__ v16bf load_tile16x32(const __bf16* __restrict__ base,
                                                int ld, int row0, int k0) {
  const int lane = threadIdx.x & 31;
  const int r    = lane & 15;
  const int hi   = lane >> 4;
  const __bf16* p = base + (size_t)(row0 + r) * ld + k0 + hi * 8;
  v8bf lo = *(const v8bf*)(p);
  v8bf hv = *(const v8bf*)(p + 16);
  v16bf out;
#pragma unroll
  for (int i = 0; i < 8; ++i) { out[i] = lo[i]; out[i + 8] = hv[i]; }
  return out;
}

// Async DMA of a 16x64 bf16 tile (2 KB) global -> LDS.  4 instructions, each
// moving 32 lanes x 16 B.  Tracked on ASYNCcnt (in-order completion).
// LDS destination address = low 32 bits of the generic __shared__ pointer.
__device__ __forceinline__ void async_copy_16x64(const __bf16* __restrict__ gsrc,
                                                 int ld, __bf16* lbuf) {
  const int lane = threadIdx.x & 31;
  const int row  = lane >> 1;          // 0..15
  const int ch0  = (lane & 1) * 4;     // 16B-chunk base: 0 or 4
#pragma unroll
  for (int it = 0; it < 4; ++it) {
    const __bf16* g = gsrc + (size_t)row * ld + (ch0 + it) * 8;
    unsigned l = (unsigned)(uintptr_t)(lbuf + row * 64 + (ch0 + it) * 8);
    asm volatile("global_load_async_to_lds_b128 %0, %1, off"
                 :: "v"(l), "v"(g) : "memory");
  }
}

// ---------------------------------------------------------------------------
// f32 -> bf16 conversion (weights)
// ---------------------------------------------------------------------------
__global__ __launch_bounds__(256) void cvt_bf16_kernel(const float* __restrict__ src,
                                                       __bf16* __restrict__ dst,
                                                       int n4) {
  int i = blockIdx.x * 256 + threadIdx.x;
  if (i < n4) {
    v4f s = ((const v4f*)src)[i];
    v4bf d;
#pragma unroll
    for (int j = 0; j < 4; ++j) d[j] = (__bf16)s[j];
    ((v4bf*)dst)[i] = d;
  }
}

// ---------------------------------------------------------------------------
// LayerNorm: one wave (32 lanes) per row of E=512.  Optional f32 and bf16 out.
// ---------------------------------------------------------------------------
__global__ __launch_bounds__(256) void ln_kernel(const float* __restrict__ x,
                                                 const float* __restrict__ w,
                                                 const float* __restrict__ b,
                                                 float* __restrict__ outF,
                                                 __bf16* __restrict__ outB,
                                                 int M) {
  const int wave = threadIdx.x >> 5;
  const int lane = threadIdx.x & 31;
  const int row  = blockIdx.x * 8 + wave;
  if (row >= M) return;

  const float* xr = x + (size_t)row * E_DIM + lane * 16;
  float v[16];
  float s = 0.f;
#pragma unroll
  for (int i = 0; i < 4; ++i) {
    v4f t = *(const v4f*)(xr + i * 4);
#pragma unroll
    for (int j = 0; j < 4; ++j) { v[i * 4 + j] = t[j]; s += t[j]; }
  }
#pragma unroll
  for (int d = 1; d < 32; d <<= 1) s += __shfl_xor(s, d, 32);
  const float mu = s * (1.f / E_DIM);

  float vs = 0.f;
#pragma unroll
  for (int i = 0; i < 16; ++i) { float d = v[i] - mu; vs += d * d; }
#pragma unroll
  for (int d = 1; d < 32; d <<= 1) vs += __shfl_xor(vs, d, 32);
  const float rs = rsqrtf(vs * (1.f / E_DIM) + 1e-5f);

  const int c0 = lane * 16;
#pragma unroll
  for (int i = 0; i < 4; ++i) {
    v4f wy = *(const v4f*)(w + c0 + i * 4);
    v4f by = *(const v4f*)(b + c0 + i * 4);
    v4f y;
    v4bf yb;
#pragma unroll
    for (int j = 0; j < 4; ++j) {
      float t = (v[i * 4 + j] - mu) * rs * wy[j] + by[j];
      y[j]  = t;
      yb[j] = (__bf16)t;
    }
    if (outF) *(v4f*)(outF + (size_t)row * E_DIM + c0 + i * 4) = y;
    if (outB) *(v4bf*)(outB + (size_t)row * E_DIM + c0 + i * 4) = yb;
  }
}

// ---------------------------------------------------------------------------
// WMMA GEMM (K = N = 512 hardcoded):
//   out[m][n] = sum_k A[m][k]*W[n][k] (+bias[n]) (+resid[m][n])
// A: [M][512] bf16 row-major.  W: [512][512] bf16 row-major.
// Outputs (any may be null): f32 row-major, bf16 row-major,
// bf16 transposed outT[n][m] (ld = M_TOT) -- lays out V as [dh][M].
// Workgroup = 8 waves = 32(M) x 256(N) tile; each wave owns 16x64.
// k-loop fully unrolled as a two-stage ping-pong pipeline: each operand set
// is refilled in place by loads (no register copies), so the next k-step's
// 5 tiles are always in flight behind the current 4 WMMAs.
// ---------------------------------------------------------------------------
__global__ __launch_bounds__(256) void gemm_kernel(const __bf16* __restrict__ A,
                                                   const __bf16* __restrict__ W,
                                                   const float* __restrict__ bias,
                                                   const float* __restrict__ resid,
                                                   float* __restrict__ outF,
                                                   __bf16* __restrict__ outB,
                                                   __bf16* __restrict__ outT) {
  constexpr int K = E_DIM;
  constexpr int N = E_DIM;
  const int wave = threadIdx.x >> 5;
  const int lane = threadIdx.x & 31;
  const int r16  = lane & 15;
  const int hi   = lane >> 4;
  const int m0   = blockIdx.x * 32 + (wave & 1) * 16;
  const int n0   = blockIdx.y * 256 + (wave >> 1) * 64;

  v8f acc[4] = {v8f{}, v8f{}, v8f{}, v8f{}};

  // stage 0 preload
  v16bf a0 = load_tile16x32(A, K, m0, 0);
  v16bf b0[4];
#pragma unroll
  for (int t = 0; t < 4; ++t) b0[t] = load_tile16x32(W, K, n0 + t * 16, 0);

#pragma unroll
  for (int k = 0; k < K; k += 64) {
    // fill stage 1 (k+32) while stage 0 computes
    v16bf a1 = load_tile16x32(A, K, m0, k + 32);
    v16bf b1[4];
#pragma unroll
    for (int t = 0; t < 4; ++t) b1[t] = load_tile16x32(W, K, n0 + t * 16, k + 32);
    if (k + 64 < K) {
      __builtin_prefetch(A + (size_t)(m0 + r16) * K + k + 64 + hi * 8, 0, 1);
      __builtin_prefetch(W + (size_t)(n0 + r16) * K + k + 64 + hi * 8, 0, 1);
    }
#pragma unroll
    for (int t = 0; t < 4; ++t) acc[t] = wmma_bf16(a0, b0[t], acc[t]);

    // refill stage 0 (k+64) while stage 1 computes
    if (k + 64 < K) {
      a0 = load_tile16x32(A, K, m0, k + 64);
#pragma unroll
      for (int t = 0; t < 4; ++t) b0[t] = load_tile16x32(W, K, n0 + t * 16, k + 64);
    }
#pragma unroll
    for (int t = 0; t < 4; ++t) acc[t] = wmma_bf16(a1, b1[t], acc[t]);
  }

  // Epilogue.  C/D layout: VGPR rr, lane -> row = m0 + rr + hi*8, col = n0+t*16+(lane&15)
#pragma unroll
  for (int t = 0; t < 4; ++t) {
    const int n  = n0 + t * 16 + r16;
    const float bv = bias ? bias[n] : 0.f;
    if (outF || outB) {
#pragma unroll
      for (int rr = 0; rr < 8; ++rr) {
        const int m = m0 + rr + hi * 8;
        float val = acc[t][rr] + bv;
        if (resid) val += resid[(size_t)m * N + n];
        if (outF) outF[(size_t)m * N + n] = val;
        if (outB) outB[(size_t)m * N + n] = (__bf16)val;
      }
    }
    if (outT) {  // rows contiguous along m in [n][m] layout -> one b128 store
      v8bf pack;
#pragma unroll
      for (int rr = 0; rr < 8; ++rr) pack[rr] = (__bf16)(acc[t][rr] + bv);
      *(v8bf*)(outT + (size_t)n * M_TOT + m0 + hi * 8) = pack;
    }
  }
}

// ---------------------------------------------------------------------------
// Flash-style attention.  Workgroup = (batch b, 16-row q tile); wave = head.
//   pass 1: online rowmax m / rowsum l over QK^T/8 (WMMA bf16)
//   pass 2: P = exp(S-m)/l ; head-mean accumulated into LDS (ds_add_f32);
//           P (bf16, restaged through LDS into A-operand layout) x V via WMMA.
// K tiles are DMA'd global->LDS with GLOBAL_LOAD_ASYNC_TO_LDS_B128, double
// buffered per wave and synchronized with s_wait_asynccnt.
// q_ws,k_ws: [M][E] bf16 row-major.  vT_ws: [E][M] bf16 ([h*64+d][m]).
// ctx_ws: [M][E] bf16.  attnW: [B][S][S] f32 (head mean).
// LDS: 64 KB accW + 8 KB P-stage + 32 KB K-stage = 104 KB (WGP has 320 KB).
// ---------------------------------------------------------------------------
__global__ __launch_bounds__(256) void attn_kernel(const __bf16* __restrict__ q_ws,
                                                   const __bf16* __restrict__ k_ws,
                                                   const __bf16* __restrict__ vT_ws,
                                                   __bf16* __restrict__ ctx_ws,
                                                   float* __restrict__ attnW) {
  __shared__ __attribute__((aligned(16))) float  accW[16 * S_LEN];           // 64 KB
  __shared__ __attribute__((aligned(16))) __bf16 stage[H_HEADS][16 * 32];    //  8 KB
  __shared__ __attribute__((aligned(16))) __bf16 kstage[H_HEADS][2][16 * 64];// 32 KB

  const int tid  = threadIdx.x;
  const int wave = tid >> 5;            // head index
  const int lane = tid & 31;
  const int r16  = lane & 15;
  const int hi   = lane >> 4;
  const int b    = blockIdx.x >> 6;     // 64 q-tiles per batch
  const int m0   = (blockIdx.x & 63) * 16;
  const int h    = wave;

  // zero the head-mean accumulator
  for (int i = tid; i < 16 * S_LEN; i += 256) accW[i] = 0.f;

  const __bf16* qbase = q_ws + (size_t)(b * S_LEN + m0) * E_DIM + h * DH;
  const __bf16* kbase = k_ws + (size_t)b * S_LEN * E_DIM + h * DH;
  const __bf16* vbase = vT_ws + (size_t)h * DH * M_TOT + (size_t)b * S_LEN;
  __bf16* kst0 = &kstage[wave][0][0];
  __bf16* kst1 = &kstage[wave][1][0];

  const v16bf qa0 = load_tile16x32(qbase, E_DIM, 0, 0);
  const v16bf qa1 = load_tile16x32(qbase, E_DIM, 0, 32);

  const float scale = 0.125f;  // 1/sqrt(64)
  float mrow[8], lrow[8];
#pragma unroll
  for (int rr = 0; rr < 8; ++rr) { mrow[rr] = -1e30f; lrow[rr] = 0.f; }

  // ---- pass 1: row stats (K streamed via async DMA, double buffered) ----
  async_copy_16x64(kbase, E_DIM, kst0);
  for (int kt = 0; kt < S_LEN / 16; ++kt) {
    __bf16* cur = (kt & 1) ? kst1 : kst0;
    __bf16* nxt = (kt & 1) ? kst0 : kst1;
    if (kt < S_LEN / 16 - 1) {
      async_copy_16x64(kbase + (size_t)(kt + 1) * 16 * E_DIM, E_DIM, nxt);
      asm volatile("s_wait_asynccnt 0x4" ::: "memory");  // current tile resident
    } else {
      asm volatile("s_wait_asynccnt 0x0" ::: "memory");
    }
    v16bf kb0 = load_tile16x32(cur, DH, 0, 0);
    v16bf kb1 = load_tile16x32(cur, DH, 0, 32);
    v8f c = v8f{};
    c = wmma_bf16(qa0, kb0, c);
    c = wmma_bf16(qa1, kb1, c);
#pragma unroll
    for (int rr = 0; rr < 8; ++rr) {
      float s = c[rr] * scale;
      float mx = s;
#pragma unroll
      for (int d = 1; d < 16; d <<= 1) mx = fmaxf(mx, __shfl_xor(mx, d, 32));
      const float nm = fmaxf(mrow[rr], mx);
      float ps = __expf(s - nm);
#pragma unroll
      for (int d = 1; d < 16; d <<= 1) ps += __shfl_xor(ps, d, 32);
      lrow[rr] = lrow[rr] * __expf(mrow[rr] - nm) + ps;
      mrow[rr] = nm;
    }
  }
  float invl[8];
#pragma unroll
  for (int rr = 0; rr < 8; ++rr) invl[rr] = 1.f / lrow[rr];

  __syncthreads();  // accW zero visible before pass-2 atomics

  // ---- pass 2: P, head-mean accumulate, CTX += P @ V ----
  v8f ctx[4] = {v8f{}, v8f{}, v8f{}, v8f{}};
  __bf16* st = &stage[wave][0];

  async_copy_16x64(kbase, E_DIM, kst0);
  for (int kt = 0; kt < S_LEN / 16; ++kt) {
    __bf16* cur = (kt & 1) ? kst1 : kst0;
    __bf16* nxt = (kt & 1) ? kst0 : kst1;
    if (kt < S_LEN / 16 - 1) {
      async_copy_16x64(kbase + (size_t)(kt + 1) * 16 * E_DIM, E_DIM, nxt);
      asm volatile("s_wait_asynccnt 0x4" ::: "memory");
    } else {
      asm volatile("s_wait_asynccnt 0x0" ::: "memory");
    }
    v16bf kb0 = load_tile16x32(cur, DH, 0, 0);
    v16bf kb1 = load_tile16x32(cur, DH, 0, 32);
    v8f c = v8f{};
    c = wmma_bf16(qa0, kb0, c);
    c = wmma_bf16(qa1, kb1, c);
#pragma unroll
    for (int rr = 0; rr < 8; ++rr) {
      const float p  = __expf(c[rr] * scale - mrow[rr]) * invl[rr];
      const int row  = rr + hi * 8;
      const int col  = (kt & 1) * 16 + r16;
      atomicAdd(&accW[row * S_LEN + kt * 16 + r16], p * (1.f / H_HEADS));
      st[row * 32 + col] = (__bf16)p;   // 16x32 staging, row-major
    }
    if (kt & 1) {  // two P tiles staged -> one K=32 PV step
      const int key0 = (kt >> 1) * 32;
      // wave-private staging: DS ops are in-order per wave; fence the counter
      asm volatile("s_wait_dscnt 0x0" ::: "memory");
      v16bf pa = load_tile16x32(st, 32, 0, 0);
#pragma unroll
      for (int t = 0; t < 4; ++t) {
        v16bf vb = load_tile16x32(vbase, M_TOT, t * 16, key0);
        ctx[t] = wmma_bf16(pa, vb, ctx[t]);
      }
    }
  }

  // CTX -> ctx_ws [M][E] bf16
  __bf16* cbase = ctx_ws + (size_t)(b * S_LEN + m0) * E_DIM + h * DH;
#pragma unroll
  for (int t = 0; t < 4; ++t)
#pragma unroll
    for (int rr = 0; rr < 8; ++rr)
      cbase[(size_t)(rr + hi * 8) * E_DIM + t * 16 + r16] = (__bf16)ctx[t][rr];

  __syncthreads();  // all heads' accW contributions done

  // head-mean -> global attn weights (layout matches accW exactly)
  float* wout = attnW + ((size_t)b * S_LEN + m0) * S_LEN;
  for (int i = tid; i < (16 * S_LEN) / 4; i += 256)
    ((v4f*)wout)[i] = ((const v4f*)accW)[i];
}

// ---------------------------------------------------------------------------
// Host orchestration
// ---------------------------------------------------------------------------
extern "C" void kernel_launch(void* const* d_in, const int* in_sizes, int n_in,
                              void* d_out, int out_size, void* d_ws, size_t ws_size,
                              hipStream_t stream) {
  const float* Zab        = (const float*)d_in[0];
  const float* Za         = (const float*)d_in[1];
  const float* ln_w       = (const float*)d_in[2];
  const float* ln_b       = (const float*)d_in[3];
  const float* in_proj_w  = (const float*)d_in[4];
  const float* in_proj_b  = (const float*)d_in[5];
  const float* out_proj_w = (const float*)d_in[6];
  const float* out_proj_b = (const float*)d_in[7];
  const float* ff_w       = (const float*)d_in[8];
  const float* ff_b       = (const float*)d_in[9];

  float* final_out = (float*)d_out;                              // [8,1024,512]
  float* attn_out  = (float*)d_out + (size_t)M_TOT * E_DIM;      // [8,1024,1024]

  char* ws = (char*)d_ws;
  size_t off = 0;
  auto alloc = [&](size_t bytes) -> void* {
    void* p = ws + off;
    off += (bytes + 255) & ~(size_t)255;
    return p;
  };

  __bf16* qin_bf  = (__bf16*)alloc((size_t)M_TOT * E_DIM * 2);
  __bf16* kvin_bf = (__bf16*)alloc((size_t)M_TOT * E_DIM * 2);
  __bf16* wqkv_bf = (__bf16*)alloc((size_t)3 * E_DIM * E_DIM * 2);
  __bf16* wout_bf = (__bf16*)alloc((size_t)E_DIM * E_DIM * 2);
  __bf16* wff_bf  = (__bf16*)alloc((size_t)E_DIM * E_DIM * 2);
  __bf16* q_bf    = (__bf16*)alloc((size_t)M_TOT * E_DIM * 2);
  __bf16* k_bf    = (__bf16*)alloc((size_t)M_TOT * E_DIM * 2);
  __bf16* vT_bf   = (__bf16*)alloc((size_t)M_TOT * E_DIM * 2);  // [E][M]
  __bf16* ctx_bf  = (__bf16*)alloc((size_t)M_TOT * E_DIM * 2);
  float*  bufX    = (float*)alloc((size_t)M_TOT * E_DIM * 4);   // x, then y
  float*  xn_f32  = (float*)alloc((size_t)M_TOT * E_DIM * 4);
  __bf16* xn_bf   = (__bf16*)alloc((size_t)M_TOT * E_DIM * 2);

  // 1) weights -> bf16
  {
    int n4;
    n4 = 3 * E_DIM * E_DIM / 4;
    cvt_bf16_kernel<<<(n4 + 255) / 256, 256, 0, stream>>>(in_proj_w, wqkv_bf, n4);
    n4 = E_DIM * E_DIM / 4;
    cvt_bf16_kernel<<<(n4 + 255) / 256, 256, 0, stream>>>(out_proj_w, wout_bf, n4);
    cvt_bf16_kernel<<<(n4 + 255) / 256, 256, 0, stream>>>(ff_w, wff_bf, n4);
  }

  // 2) input layernorms (bf16 only needed)
  ln_kernel<<<M_TOT / 8, 256, 0, stream>>>(Zab, ln_w, ln_b, nullptr, qin_bf, M_TOT);
  ln_kernel<<<M_TOT / 8, 256, 0, stream>>>(Za,  ln_w, ln_b, nullptr, kvin_bf, M_TOT);

  // 3) QKV projections (N = 512 each; V stored transposed [dh-major][M])
  dim3 ggrid(M_TOT / 32, E_DIM / 256);
  gemm_kernel<<<ggrid, 256, 0, stream>>>(qin_bf,  wqkv_bf,
                                         in_proj_b, nullptr,
                                         nullptr, q_bf, nullptr);
  gemm_kernel<<<ggrid, 256, 0, stream>>>(kvin_bf, wqkv_bf + (size_t)E_DIM * E_DIM,
                                         in_proj_b + E_DIM, nullptr,
                                         nullptr, k_bf, nullptr);
  gemm_kernel<<<ggrid, 256, 0, stream>>>(kvin_bf, wqkv_bf + (size_t)2 * E_DIM * E_DIM,
                                         in_proj_b + 2 * E_DIM, nullptr,
                                         nullptr, nullptr, vT_bf);

  // 4) attention (flash two-pass, async-DMA K staging, head-mean to d_out)
  attn_kernel<<<B_BATCH * (S_LEN / 16), 256, 0, stream>>>(q_bf, k_bf, vT_bf,
                                                          ctx_bf, attn_out);

  // 5) out-proj + residual -> x (f32)
  gemm_kernel<<<ggrid, 256, 0, stream>>>(ctx_bf, wout_bf,
                                         out_proj_b, Zab,
                                         bufX, nullptr, nullptr);

  // 6) x_n = LN(x)  (f32 + bf16)
  ln_kernel<<<M_TOT / 8, 256, 0, stream>>>(bufX, ln_w, ln_b, xn_f32, xn_bf, M_TOT);

  // 7) y = x_n @ ff_w^T + ff_b + x_n   (reuse bufX for y)
  gemm_kernel<<<ggrid, 256, 0, stream>>>(xn_bf, wff_bf,
                                         ff_b, xn_f32,
                                         bufX, nullptr, nullptr);

  // 8) final = LN(y) -> d_out
  ln_kernel<<<M_TOT / 8, 256, 0, stream>>>(bufX, ln_w, ln_b, final_out, nullptr, M_TOT);
}
